// Block_55044300865696
// MI455X (gfx1250) — matrix operand order repaired and checked by reading, not compiled, not run on hardware
//
#include <hip/hip_runtime.h>
#include <math.h>

// ---------------- problem constants ----------------
#define BB 2
#define TT 2048
#define CC 1024
#define HH 16
#define HD 64
#define MM (BB * TT)        // 4096 rows

typedef unsigned short u16;
typedef __bf16 bf16_t;
typedef __attribute__((ext_vector_type(16))) bf16_t v16bf;
typedef __attribute__((ext_vector_type(8)))  float   v8f;
typedef __attribute__((ext_vector_type(4)))  u16     u16x4;
typedef __attribute__((ext_vector_type(4)))  unsigned int v4u;
typedef __attribute__((ext_vector_type(8)))  int     v8i;
typedef __attribute__((ext_vector_type(4)))  int     v4i;

union FragBF { v16bf v; uint4 q[2]; };

__device__ __forceinline__ u16 f2bf(float f) {
  unsigned u = __float_as_uint(f);
  unsigned r = u + 0x7FFFu + ((u >> 16) & 1u);   // RNE
  return (u16)(r >> 16);
}

__device__ __forceinline__ v8f wmma_bf16(const FragBF& a, const FragBF& b, v8f c) {
  return __builtin_amdgcn_wmma_f32_16x16x32_bf16(
      false, a.v, false, b.v, (short)0, c, false, false);
}

__device__ __forceinline__ float gelu_f(float x) {
  return 0.5f * x * (1.0f + tanhf(0.7978845608028654f * (x + 0.044715f * x * x * x)));
}

// ---- TDM: DMA a [128 rows x 32 bf16] tile into LDS with 80B padded rows ----
// D# layout per CDNA5 ISA.  6-arg builtin variant.
__device__ __forceinline__ void tdm_load_tile(unsigned lds_off, const u16* gsrc,
                                              int Kdim) {
  unsigned long long ga = (unsigned long long)(uintptr_t)gsrc;
  v4u g0;
  g0.x = 1u;                                            // count=1, user mode
  g0.y = lds_off;                                       // lds_addr (bytes)
  g0.z = (unsigned)(ga & 0xFFFFFFFFu);                  // global_addr lo
  g0.w = (unsigned)((ga >> 32) & 0x01FFFFFFu) | (2u << 30);  // addr hi | type=2
  v8i g1;
  // data_size=1 (2B) | pad_enable | pad_interval=3 (16 DW) | pad_amount=3 (4 DW)
  g1[0] = (int)((1u << 16) | (1u << 20) | (3u << 22) | (3u << 25));
  unsigned d0 = (unsigned)Kdim;                         // tensor_dim0 (elements)
  unsigned d1 = (unsigned)MM;                           // tensor_dim1 (rows)
  g1[1] = (int)((d0 & 0xFFFFu) << 16);                  // abar_addr=0 | dim0 lo16
  g1[2] = (int)((d0 >> 16) | ((d1 & 0xFFFFu) << 16));   // dim0 hi16 | dim1 lo16
  g1[3] = (int)((d1 >> 16) | (32u << 16));              // dim1 hi16 | tile_dim0=32
  g1[4] = 128;                                          // tile_dim1=128 | tile_dim2=0
  g1[5] = Kdim;                                         // dim0_stride lo32 (elements)
  g1[6] = 0;                                            // dim0_stride hi16 | dim1_stride lo
  g1[7] = 0;
  v4i z4 = {0, 0, 0, 0};
  v8i z8 = {0, 0, 0, 0, 0, 0, 0, 0};
  __builtin_amdgcn_tensor_load_to_lds(g0, g1, z4, z4, z8, 0);
}

// ---------------- fp32 -> bf16 convert ----------------
__global__ __launch_bounds__(256) void cvt_bf16_kernel(const float* __restrict__ in,
                                                       u16* __restrict__ out, int n) {
  int i = blockIdx.x * 256 + threadIdx.x;
  if (i < n) out[i] = f2bf(in[i]);
}

// ---------------- LayerNorm (fp32 in, bf16 out), one row per block ----------------
__global__ __launch_bounds__(256) void ln_kernel(const float* __restrict__ x,
                                                 const float* __restrict__ g,
                                                 const float* __restrict__ b,
                                                 u16* __restrict__ out) {
  const int row  = blockIdx.x;
  const int tid  = threadIdx.x;
  const int lane = tid & 31, wave = tid >> 5;
  const float4 xv = ((const float4*)(x + (size_t)row * CC))[tid];
  float s1 = xv.x + xv.y + xv.z + xv.w;
  float s2 = xv.x * xv.x + xv.y * xv.y + xv.z * xv.z + xv.w * xv.w;
#pragma unroll
  for (int m = 1; m < 32; m <<= 1) {
    s1 += __shfl_xor(s1, m, 32);
    s2 += __shfl_xor(s2, m, 32);
  }
  __shared__ float w1[8], w2[8];
  __shared__ float smu, srs;
  if (lane == 0) { w1[wave] = s1; w2[wave] = s2; }
  __syncthreads();
  if (tid == 0) {
    float a = 0.f, c = 0.f;
#pragma unroll
    for (int j = 0; j < 8; ++j) { a += w1[j]; c += w2[j]; }
    float mu  = a * (1.0f / CC);
    float var = c * (1.0f / CC) - mu * mu;
    smu = mu;
    srs = rsqrtf(var + 1e-5f);
  }
  __syncthreads();
  const float4 gv = ((const float4*)g)[tid];
  const float4 bv = ((const float4*)b)[tid];
  u16x4 ov;
  ov.x = f2bf((xv.x - smu) * srs * gv.x + bv.x);
  ov.y = f2bf((xv.y - smu) * srs * gv.y + bv.y);
  ov.z = f2bf((xv.z - smu) * srs * gv.z + bv.z);
  ov.w = f2bf((xv.w - smu) * srs * gv.w + bv.w);
  ((u16x4*)(out + (size_t)row * CC))[tid] = ov;
}

// ---------------- qkv [M,3C] bf16 -> Q/K [BH,T,64], Vt [BH,64,T] ----------------
__global__ __launch_bounds__(256) void qkv_reorder_kernel(const u16* __restrict__ qkv,
                                                          u16* __restrict__ q,
                                                          u16* __restrict__ k,
                                                          u16* __restrict__ vt) {
  size_t idx = (size_t)blockIdx.x * 256 + threadIdx.x;     // over M*C
  int c  = (int)(idx % CC);
  size_t row = idx / CC;
  int b = (int)(row / TT), t = (int)(row % TT);
  int h = c >> 6, d = c & 63;
  size_t bh = (size_t)b * HH + h;
  const u16* src = qkv + row * (3 * CC);
  q[(bh * TT + t) * HD + d]       = src[c];
  k[(bh * TT + t) * HD + d]       = src[CC + c];
  vt[(bh * HD + d) * TT + t]      = src[2 * CC + c];
}

// ---------------- GEMM: out[M,N] = A[M,K](bf16) * Bw[N,K]^T(bf16) + bias (+epilogue) ----
// 128x128x32 tile, 256 threads = 8 waves (2 N x 4 M), wave tile 32x64.
// Global->LDS staging via Tensor Data Mover (double buffered, TENSORcnt).
template <int GELU, int RESID, int OUTBF>
__global__ __launch_bounds__(256) void gemm_kernel(const u16* __restrict__ A,
                                                   const u16* __restrict__ Bw,
                                                   const float* __restrict__ bias,
                                                   const float* __restrict__ resid,
                                                   float* __restrict__ outF,
                                                   u16* __restrict__ outB,
                                                   int Ndim, int Kdim) {
  __shared__ u16 lA[2][128 * 40];   // rows padded to 80B by TDM -> conflict-free b128 reads
  __shared__ u16 lB[2][128 * 40];
  const int tid  = threadIdx.x;
  const int lane = tid & 31, wave = tid >> 5;
  const int half = (lane >> 4) & 1, r = lane & 15;
  const int mw = (wave & 3) * 32;
  const int nw = (wave >> 2) * 64;
  const int m0 = blockIdx.y * 128;
  const int n0 = blockIdx.x * 128;

  const unsigned ldsA0 = (unsigned)(uintptr_t)(&lA[0][0]);
  const unsigned ldsA1 = (unsigned)(uintptr_t)(&lA[1][0]);
  const unsigned ldsB0 = (unsigned)(uintptr_t)(&lB[0][0]);
  const unsigned ldsB1 = (unsigned)(uintptr_t)(&lB[1][0]);

  v8f acc[2][4];
#pragma unroll
  for (int mi = 0; mi < 2; ++mi)
#pragma unroll
    for (int ni = 0; ni < 4; ++ni) acc[mi][ni] = (v8f){0, 0, 0, 0, 0, 0, 0, 0};

  const int KT = Kdim >> 5;
  if (wave == 0) {                         // prologue: DMA tile 0 into buf 0
    tdm_load_tile(ldsA0, A  + (size_t)m0 * Kdim, Kdim);
    tdm_load_tile(ldsB0, Bw + (size_t)n0 * Kdim, Kdim);
  }

  for (int kt = 0; kt < KT; ++kt) {
    const int buf = kt & 1;
    if (wave == 0) {
      if (kt + 1 < KT) {                   // DMA next tile into other buffer
        const int k1 = (kt + 1) * 32;
        tdm_load_tile(buf ? ldsA0 : ldsA1, A  + (size_t)m0 * Kdim + k1, Kdim);
        tdm_load_tile(buf ? ldsB0 : ldsB1, Bw + (size_t)n0 * Kdim + k1, Kdim);
        __builtin_amdgcn_s_wait_tensorcnt(2);  // in-order: current tile has landed
      } else {
        __builtin_amdgcn_s_wait_tensorcnt(0);
      }
    }
    __syncthreads();                       // release all waves to consume buf

    FragBF af[2], bfm[4];
#pragma unroll
    for (int mi = 0; mi < 2; ++mi) {
      const u16* p = &lA[buf][(mw + mi * 16 + r) * 40];
      af[mi].q[0] = *(const uint4*)(p + half * 8);        // A: K {0..7|8..15}
      af[mi].q[1] = *(const uint4*)(p + 16 + half * 8);   //    K {16..23|24..31}
    }
#pragma unroll
    for (int ni = 0; ni < 4; ++ni) {
      const u16* p = &lB[buf][(nw + ni * 16 + r) * 40];
      bfm[ni].q[0] = *(const uint4*)(p + half * 8);       // B: K {0..7|8..15}
      bfm[ni].q[1] = *(const uint4*)(p + 16 + half * 8);  //    K {16..23|24..31}
    }
#pragma unroll
    for (int mi = 0; mi < 2; ++mi)
#pragma unroll
      for (int ni = 0; ni < 4; ++ni)
        acc[mi][ni] = wmma_bf16(af[mi], bfm[ni], acc[mi][ni]);

    __syncthreads();                       // all reads of buf done before it is re-filled
  }

  // epilogue
#pragma unroll
  for (int mi = 0; mi < 2; ++mi) {
#pragma unroll
    for (int ni = 0; ni < 4; ++ni) {
#pragma unroll
      for (int i = 0; i < 8; ++i) {
        int rg = m0 + mw + mi * 16 + i + half * 8;
        int cg = n0 + nw + ni * 16 + r;
        float v = acc[mi][ni][i] + bias[cg];
        if (GELU) v = gelu_f(v);
        if (RESID) v += resid[(size_t)rg * Ndim + cg];
        if (OUTBF) outB[(size_t)rg * Ndim + cg] = f2bf(v);
        else       outF[(size_t)rg * Ndim + cg] = v;
      }
    }
  }
}

// ---------------- flash attention: 128 q rows per block, 16 per wave ----------------
__global__ __launch_bounds__(256) void attn_kernel(const u16* __restrict__ Q,
                                                   const u16* __restrict__ K,
                                                   const u16* __restrict__ Vt,
                                                   u16* __restrict__ Y) {
  __shared__ u16 ldsP[8][16 * 40];   // per-wave scratch: 16x32 P tile, 80B row stride
  const int lane = threadIdx.x & 31;
  const int wave = threadIdx.x >> 5;
  const int half = (lane >> 4) & 1, r = lane & 15;
  const int bh = blockIdx.y;                 // b*H + h
  const int b  = bh >> 4, h = bh & 15;
  const int q0 = blockIdx.x * 128 + wave * 16;
  const float scale = 0.125f;                // 1/sqrt(64)

  // Q fragments (A layout), resident for whole kernel
  FragBF qf[2];
  {
    const u16* qrow = Q + ((size_t)bh * TT + (q0 + r)) * HD;
    qf[0].q[0] = *(const uint4*)(qrow + half * 8);
    qf[0].q[1] = *(const uint4*)(qrow + 16 + half * 8);
    qf[1].q[0] = *(const uint4*)(qrow + 32 + half * 8);
    qf[1].q[1] = *(const uint4*)(qrow + 48 + half * 8);
  }

  v8f o[4];
#pragma unroll
  for (int d = 0; d < 4; ++d) o[d] = (v8f){0, 0, 0, 0, 0, 0, 0, 0};
  float mrun[8], lrun[8];
#pragma unroll
  for (int i = 0; i < 8; ++i) { mrun[i] = -1e30f; lrun[i] = 0.f; }

  const int ntiles = (q0 + 16 + 31) >> 5;    // causal: kv <= q0+15
  for (int nt = 0; nt < ntiles; ++nt) {
    const int n0 = nt * 32;
    v8f s[2];
    s[0] = (v8f){0, 0, 0, 0, 0, 0, 0, 0};
    s[1] = (v8f){0, 0, 0, 0, 0, 0, 0, 0};
#pragma unroll
    for (int nsub = 0; nsub < 2; ++nsub) {
      const u16* krow = K + ((size_t)bh * TT + (n0 + nsub * 16 + r)) * HD;
#pragma unroll
      for (int ks = 0; ks < 2; ++ks) {
        FragBF kf;                                           // B layout
        kf.q[0] = *(const uint4*)(krow + ks * 32 + half * 8);
        kf.q[1] = *(const uint4*)(krow + ks * 32 + 16 + half * 8);
        s[nsub] = wmma_bf16(qf[ks], kf, s[nsub]);
      }
    }
    // online softmax update (per row slot), then P -> LDS (bf16)
#pragma unroll
    for (int i = 0; i < 8; ++i) {
      const int qm = q0 + i + half * 8;
      float s0 = s[0][i] * scale;
      float s1 = s[1][i] * scale;
      if (n0 + r > qm)      s0 = -1e30f;
      if (n0 + 16 + r > qm) s1 = -1e30f;
      float tm = fmaxf(s0, s1);
#pragma unroll
      for (int m = 1; m < 16; m <<= 1) tm = fmaxf(tm, __shfl_xor(tm, m, 32));
      const float mnew = fmaxf(mrun[i], tm);
      const float a    = __expf(mrun[i] - mnew);
      const float p0   = __expf(s0 - mnew);
      const float p1   = __expf(s1 - mnew);
      float ts = p0 + p1;
#pragma unroll
      for (int m = 1; m < 16; m <<= 1) ts += __shfl_xor(ts, m, 32);
      lrun[i] = lrun[i] * a + ts;
      mrun[i] = mnew;
#pragma unroll
      for (int d = 0; d < 4; ++d) o[d][i] *= a;
      ldsP[wave][(i + half * 8) * 40 + r]      = f2bf(p0);
      ldsP[wave][(i + half * 8) * 40 + 16 + r] = f2bf(p1);
    }
    // wave-private LDS RAW: same-wave DS ops are in-order; wait DScnt, no barrier
    asm volatile("s_wait_dscnt 0" ::: "memory");

    FragBF pf;                                               // A layout reload of P
    pf.q[0] = *(const uint4*)(&ldsP[wave][r * 40 + half * 8]);
    pf.q[1] = *(const uint4*)(&ldsP[wave][r * 40 + 16 + half * 8]);
#pragma unroll
    for (int d = 0; d < 4; ++d) {
      const u16* vrow = Vt + ((size_t)bh * HD + d * 16 + r) * TT + n0;
      FragBF vf;                                             // B layout (table K order)
      vf.q[0] = *(const uint4*)(vrow + half * 8);
      vf.q[1] = *(const uint4*)(vrow + 16 + half * 8);
      o[d] = wmma_bf16(pf, vf, o[d]);
    }
  }

  // normalize + write y as [M, C] bf16 (col = h*64 + d)
#pragma unroll
  for (int i = 0; i < 8; ++i) {
    const float inv = 1.0f / lrun[i];
    const int t = q0 + i + half * 8;
    u16* yrow = Y + ((size_t)b * TT + t) * CC + h * HD;
#pragma unroll
    for (int d = 0; d < 4; ++d) yrow[d * 16 + r] = f2bf(o[d][i] * inv);
  }
}

// ---------------- host-side orchestration ----------------
extern "C" void kernel_launch(void* const* d_in, const int* in_sizes, int n_in,
                              void* d_out, int out_size, void* d_ws, size_t ws_size,
                              hipStream_t stream) {
  const float* x       = (const float*)d_in[0];
  const float* ln1_g   = (const float*)d_in[1];
  const float* ln1_b   = (const float*)d_in[2];
  const float* ln2_g   = (const float*)d_in[3];
  const float* ln2_b   = (const float*)d_in[4];
  const float* W_attn  = (const float*)d_in[5];
  const float* b_attn  = (const float*)d_in[6];
  const float* W_aproj = (const float*)d_in[7];
  const float* b_aproj = (const float*)d_in[8];
  const float* W_fc    = (const float*)d_in[9];
  const float* b_fc    = (const float*)d_in[10];
  const float* W_mproj = (const float*)d_in[11];
  const float* b_mproj = (const float*)d_in[12];

  char* base = (char*)d_ws;
  size_t off = 0;
  auto alloc = [&](size_t bytes) -> char* {
    char* p = base + off;
    off += (bytes + 255) & ~(size_t)255;
    return p;
  };
  u16*   wattn  = (u16*)alloc((size_t)3 * CC * CC * 2);
  u16*   waproj = (u16*)alloc((size_t)CC * CC * 2);
  u16*   wfc    = (u16*)alloc((size_t)4 * CC * CC * 2);
  u16*   wmproj = (u16*)alloc((size_t)4 * CC * CC * 2);
  u16*   hbuf   = (u16*)alloc((size_t)MM * CC * 2);
  u16*   qkv    = (u16*)alloc((size_t)MM * 3 * CC * 2);
  u16*   qb     = (u16*)alloc((size_t)MM * CC * 2);
  u16*   kb     = (u16*)alloc((size_t)MM * CC * 2);
  u16*   vtb    = (u16*)alloc((size_t)MM * CC * 2);
  u16*   ybuf   = (u16*)alloc((size_t)MM * CC * 2);
  float* x1     = (float*)alloc((size_t)MM * CC * 4);
  u16*   h2buf  = (u16*)alloc((size_t)MM * CC * 2);
  u16*   mbuf   = (u16*)alloc((size_t)MM * 4 * CC * 2);
  (void)ws_size; (void)in_sizes; (void)n_in; (void)out_size;

  // weights fp32 -> bf16
  auto cvt = [&](const float* src, u16* dst, int n) {
    cvt_bf16_kernel<<<(n + 255) / 256, 256, 0, stream>>>(src, dst, n);
  };
  cvt(W_attn,  wattn,  3 * CC * CC);
  cvt(W_aproj, waproj, CC * CC);
  cvt(W_fc,    wfc,    4 * CC * CC);
  cvt(W_mproj, wmproj, 4 * CC * CC);

  // LN1
  ln_kernel<<<MM, 256, 0, stream>>>(x, ln1_g, ln1_b, hbuf);

  // qkv = h @ W_attn^T + b_attn   [4096 x 3072]
  gemm_kernel<0, 0, 1><<<dim3(3 * CC / 128, MM / 128), 256, 0, stream>>>(
      hbuf, wattn, b_attn, nullptr, nullptr, qkv, 3 * CC, CC);

  // split/reorder heads
  qkv_reorder_kernel<<<(MM * CC) / 256, 256, 0, stream>>>(qkv, qb, kb, vtb);

  // causal flash attention -> y [4096 x 1024] bf16
  attn_kernel<<<dim3(TT / 128, BB * HH), 256, 0, stream>>>(qb, kb, vtb, ybuf);

  // x1 = x + y @ W_aproj^T + b_aproj   (fp32)
  gemm_kernel<0, 1, 0><<<dim3(CC / 128, MM / 128), 256, 0, stream>>>(
      ybuf, waproj, b_aproj, x, x1, nullptr, CC, CC);

  // LN2
  ln_kernel<<<MM, 256, 0, stream>>>(x1, ln2_g, ln2_b, h2buf);

  // m = gelu(h2 @ W_fc^T + b_fc)   [4096 x 4096] bf16
  gemm_kernel<1, 0, 1><<<dim3(4 * CC / 128, MM / 128), 256, 0, stream>>>(
      h2buf, wfc, b_fc, nullptr, nullptr, mbuf, 4 * CC, CC);

  // out = x1 + m @ W_mproj^T + b_mproj   (fp32)
  gemm_kernel<0, 1, 0><<<dim3(CC / 128, MM / 128), 256, 0, stream>>>(
      mbuf, wmproj, b_mproj, x1, (float*)d_out, nullptr, CC, 4 * CC);
}
